// GNNEncoder_4398046511958
// MI455X (gfx1250) — matrix operand order repaired and checked by reading, not compiled, not run on hardware
//
#include <hip/hip_runtime.h>
#include <hip/hip_bf16.h>
#include <stdint.h>

// Problem constants (from reference)
#define E_EDGES   640000
#define NN_NODES  20000
#define NG_GRAPHS 64
#define NF 128
#define EF 128
#define GD 128
#define HD 256

typedef __bf16 bf16_t;
typedef bf16_t   v16bf __attribute__((ext_vector_type(16)));
typedef float    v8f   __attribute__((ext_vector_type(8)));
typedef unsigned v4u   __attribute__((ext_vector_type(4)));
typedef unsigned v8u   __attribute__((ext_vector_type(8)));

static __device__ __forceinline__ unsigned short f2bf(float f) {
  union { float f; unsigned u; } v; v.f = f;
  unsigned r = v.u + 0x7FFFu + ((v.u >> 16) & 1u);   // round-to-nearest-even
  return (unsigned short)(r >> 16);
}
static __device__ __forceinline__ unsigned packbf2(float a, float b) {
  return (unsigned)f2bf(a) | ((unsigned)f2bf(b) << 16);
}

// ---------------------------------------------------------------------------
// Pack fp32 torch-layout weight W[N,K] into WMMA bf16 B-operand fragments:
// layout [kt][nt][lane][8 x u32]; lanes 0-15 hold K=kt*32+0..15 of col n,
// lanes 16-31 hold K=kt*32+16..31 (contiguous k order, matching the A-fragment LDS reads).
// ---------------------------------------------------------------------------
__global__ void k_pack_w(const float* __restrict__ W, unsigned* __restrict__ Bp,
                         int K, int N) {
  int idx = blockIdx.x * blockDim.x + threadIdx.x;
  int total = (K >> 5) * (N >> 4) * 32;
  if (idx >= total) return;
  int lane  = idx & 31;
  int ktnt  = idx >> 5;
  int nt16  = N >> 4;
  int nt    = ktnt % nt16;
  int kt    = ktnt / nt16;
  int n     = nt * 16 + (lane & 15);
  int kbase = kt * 32 + ((lane & 16) ? 16 : 0);
  unsigned* dst = Bp + (size_t)idx * 8;
  const float* w = W + (size_t)n * K + kbase;
#pragma unroll
  for (int j = 0; j < 8; ++j) dst[j] = packbf2(w[2 * j], w[2 * j + 1]);
}

// ---------------------------------------------------------------------------
// TDM: DMA one 128x32 bf16 A-panel (tile) into LDS with hardware padding.
//  - tile_dim0 = 32 elems (64B rows), tile_dim1 = 128 rows
//  - pad_enable, pad_interval code 3 (=16 DWORDs), pad_amount code 3 (=4 DW)
//    -> LDS row pitch 64B + 16B = 80B (APITCH), matching the ds_load layout
//  - tensor_dim1 = rows remaining -> hardware zero-fills OOB rows
// Issued by one wave per block; tracked with TENSORcnt.
// ---------------------------------------------------------------------------
static __device__ __forceinline__ void tdm_load_panel(
    const unsigned short* __restrict__ gsrc, unsigned lds_addr, int K,
    int rowsRem) {
  unsigned long long ga = (unsigned long long)(uintptr_t)gsrc;
  v4u g0;
  g0.x = 1u;                                   // count=1, user descriptor
  g0.y = lds_addr;                             // LDS byte address
  g0.z = (unsigned)ga;                         // global_addr[31:0]
  g0.w = (unsigned)((ga >> 32) & 0x01FFFFFFull) | (2u << 30);  // type=2
  v8u g1;
  g1.s0 = (1u << 16)      // data_size = 1 -> 2 bytes
        | (1u << 20)      // pad_enable
        | (3u << 22)      // pad_interval code 3 -> every 16 DWORDs
        | (3u << 25);     // pad_amount  code 3 -> 4 DWORDs (16B)
  g1.s1 = ((unsigned)K & 0xFFFFu) << 16;                 // tensor_dim0 lo16
  g1.s2 = (((unsigned)K >> 16) & 0xFFFFu)                // tensor_dim0 hi16
        | (((unsigned)rowsRem & 0xFFFFu) << 16);         // tensor_dim1 lo16
  g1.s3 = (((unsigned)rowsRem >> 16) & 0xFFFFu)          // tensor_dim1 hi16
        | (32u << 16);                                   // tile_dim0 = 32
  g1.s4 = 128u;                                          // tile_dim1 = 128
  g1.s5 = (unsigned)K;                                   // dim0_stride[31:0]
  g1.s6 = 0u;                                            // stride hi, dim1_stride lo
  g1.s7 = 0u;
  asm volatile("tensor_load_to_lds %0, %1" ::"s"(g0), "s"(g1) : "memory");
}

// ---------------------------------------------------------------------------
// bf16 WMMA GEMM: C[M,N] = A[M,K] * Bpacked + bias, optional ReLU.
// Block: 256 threads (8 wave32), tile 128x64; each wave: 32x32 (4 WMMA accs).
// A panels streamed by the Tensor Data Mover into double-buffered LDS.
// ---------------------------------------------------------------------------
#define BM 128
#define BN 64
#define APITCH 40                 // u16 per LDS A row (80B, TDM-padded)
#define PANEL_U16 (BM * APITCH)   // 5120 u16 = 10240B per buffer

__global__ __launch_bounds__(256)
void k_gemm_bf16(const unsigned short* __restrict__ A,
                 const unsigned* __restrict__ Bp,
                 const float* __restrict__ bias,
                 float* __restrict__ Cf, unsigned short* __restrict__ Cbf,
                 int M, int N, int K, int relu) {
  __shared__ __align__(16) unsigned short sA[2 * PANEL_U16];
  const int tid  = threadIdx.x;
  const int lane = tid & 31;
  const int w    = tid >> 5;
  const int wm   = w & 3;   // 0..3 : which 32-row strip
  const int wn   = w >> 2;  // 0..1 : which 32-col strip
  const int rowBase = blockIdx.x * BM;
  const int colBase = blockIdx.y * BN;
  const int nt16 = N >> 4;
  const int nk   = K >> 5;
  const int l15  = lane & 15;
  const int hi   = (lane >> 4) & 1;
  const int rowsRem = M - rowBase;

  const unsigned ldsBase = (unsigned)(uintptr_t)(void*)sA;  // LDS byte offset
  const unsigned short* Arow = A + (size_t)rowBase * K;

  const v8f vzero = {0.f, 0.f, 0.f, 0.f, 0.f, 0.f, 0.f, 0.f};
  v8f acc[2][2];
#pragma unroll
  for (int i = 0; i < 2; ++i)
#pragma unroll
    for (int j = 0; j < 2; ++j) acc[i][j] = vzero;

  // Prologue: DMA panel kt=0 into buffer 0
  if (tid < 32) {
    tdm_load_panel(Arow, ldsBase, K, rowsRem);
    __builtin_amdgcn_s_wait_tensorcnt(0);
  }
  __syncthreads();

  for (int kt = 0; kt < nk; ++kt) {
    // Issue DMA for next panel into the other buffer (overlaps compute);
    // that buffer was last read in iteration kt-1 (fenced by the barrier).
    if (kt + 1 < nk && tid < 32)
      tdm_load_panel(Arow + (kt + 1) * 32,
                     ldsBase + (unsigned)(((kt + 1) & 1) * (PANEL_U16 * 2)),
                     K, rowsRem);

    const unsigned short* sbuf = sA + (kt & 1) * PANEL_U16;

    // B fragments straight from packed global (L2-resident weights)
    union { uint4 q[2]; v16bf v; } bf[2];
#pragma unroll
    for (int tj = 0; tj < 2; ++tj) {
      int nt = (colBase >> 4) + wn * 2 + tj;
      const uint4* bp =
          (const uint4*)(Bp + (((size_t)kt * nt16 + nt) * 32 + lane) * 8);
      bf[tj].q[0] = bp[0];
      bf[tj].q[1] = bp[1];
    }
    // A fragments from LDS per 16-bit A 16x32 lane layout
    union { uint4 q[2]; v16bf v; } af[2];
#pragma unroll
    for (int ti = 0; ti < 2; ++ti) {
      int m = wm * 32 + ti * 16 + l15;
      const char* base = (const char*)&sbuf[m * APITCH];
      af[ti].q[0] = *(const uint4*)(base + hi * 32);
      af[ti].q[1] = *(const uint4*)(base + hi * 32 + 16);
    }
#pragma unroll
    for (int ti = 0; ti < 2; ++ti)
#pragma unroll
      for (int tj = 0; tj < 2; ++tj)
        acc[ti][tj] = __builtin_amdgcn_wmma_f32_16x16x32_bf16(
            false, af[ti].v, false, bf[tj].v, (short)0, acc[ti][tj],
            false, false);

    // Next panel has landed; publish to all waves
    if (tid < 32) __builtin_amdgcn_s_wait_tensorcnt(0);
    __syncthreads();
  }

  // Epilogue: bias (+ReLU), write f32 and/or bf16
#pragma unroll
  for (int ti = 0; ti < 2; ++ti) {
#pragma unroll
    for (int tj = 0; tj < 2; ++tj) {
      int gc = colBase + wn * 32 + tj * 16 + l15;
      float b = bias ? bias[gc] : 0.0f;
#pragma unroll
      for (int r = 0; r < 8; ++r) {
        int gr = rowBase + wm * 32 + ti * 16 + r + hi * 8;
        float v = acc[ti][tj][r] + b;
        if (relu) v = fmaxf(v, 0.0f);
        if (gr < M) {
          size_t o = (size_t)gr * N + gc;
          if (Cf)  Cf[o]  = v;
          if (Cbf) Cbf[o] = f2bf(v);
        }
      }
    }
  }
}

// ---------------------------------------------------------------------------
// Elementwise / reduction helpers
// ---------------------------------------------------------------------------
__global__ void k_zero(float* __restrict__ p, size_t n) {
  size_t i = (size_t)blockIdx.x * blockDim.x + threadIdx.x;
  if (i < n) p[i] = 0.0f;
}

// BatchNorm column sums/sumsq, block = C threads, chunked rows + atomics
__global__ void k_colstats(const float* __restrict__ X, float* __restrict__ st,
                           int M, int C, int rowsPerBlock) {
  int t  = threadIdx.x;
  int r0 = blockIdx.x * rowsPerBlock;
  float s = 0.f, q = 0.f;
  for (int r = 0; r < rowsPerBlock; ++r) {
    int rr = r0 + r;
    if (rr >= M) break;
    float v = X[(size_t)rr * C + t];
    s += v; q += v * v;
  }
  atomicAdd(&st[t], s);
  atomicAdd(&st[C + t], q);
}

__global__ void k_bn_act_bf16(const float* __restrict__ X,
                              const float* __restrict__ st,
                              const float* __restrict__ g,
                              const float* __restrict__ beta,
                              unsigned short* __restrict__ Y, int M, int C) {
  size_t i = (size_t)blockIdx.x * blockDim.x + threadIdx.x;
  if (i >= (size_t)M * C) return;
  int c = (int)(i % C);
  float im  = 1.0f / (float)M;
  float mu  = st[c] * im;
  float var = st[C + c] * im - mu * mu;
  float y   = (X[i] - mu) * rsqrtf(var + 1e-5f) * g[c] + beta[c];
  Y[i] = f2bf(fmaxf(y, 0.0f));
}

__global__ void k_scatter_rows(const float* __restrict__ H,
                               const int* __restrict__ idx,
                               float* __restrict__ acc,
                               float* __restrict__ cnt, int C) {
  int e = blockIdx.x, t = threadIdx.x;
  int d = idx[e];
  atomicAdd(&acc[(size_t)d * C + t], H[(size_t)e * C + t]);
  if (t == 0) atomicAdd(&cnt[d], 1.0f);
}

__global__ void k_div_rows(float* __restrict__ acc,
                           const float* __restrict__ cnt, int C) {
  int n = blockIdx.x, t = threadIdx.x;
  acc[(size_t)n * C + t] /= fmaxf(cnt[n], 1.0f);
}

// ---------------------------------------------------------------------------
// Gather/concat builders (fp32 sources -> packed bf16 pairs)
// ---------------------------------------------------------------------------
__global__ void k_build_Ae(const float* __restrict__ x, const float* __restrict__ ea,
                           const float* __restrict__ u, const int* __restrict__ ei,
                           const int* __restrict__ batch, unsigned* __restrict__ out,
                           int E) {
  int e = blockIdx.x, t = threadIdx.x;  // 256 thr, 512 cols
  int c0 = t * 2, seg = c0 >> 7, off = c0 & 127;
  const float* src;
  if (seg == 0)      src = x  + (size_t)ei[e] * NF;
  else if (seg == 1) src = x  + (size_t)ei[E + e] * NF;
  else if (seg == 2) src = ea + (size_t)e * EF;
  else               src = u  + (size_t)batch[ei[e]] * GD;
  out[(size_t)e * 256 + t] = packbf2(src[off], src[off + 1]);
}

__global__ void k_build_An1(const float* __restrict__ x, const float* __restrict__ en,
                            const int* __restrict__ ei, unsigned* __restrict__ out,
                            int E) {
  int e = blockIdx.x, t = threadIdx.x;  // 128 thr, 256 cols
  int c0 = t * 2, seg = c0 >> 7, off = c0 & 127;
  const float* src = (seg == 0) ? x + (size_t)ei[e] * NF : en + (size_t)e * EF;
  out[(size_t)e * 128 + t] = packbf2(src[off], src[off + 1]);
}

__global__ void k_build_An2(const float* __restrict__ x, const float* __restrict__ agg,
                            const float* __restrict__ u, const int* __restrict__ batch,
                            unsigned* __restrict__ out) {
  int n = blockIdx.x, t = threadIdx.x;  // 256 thr, 512 cols
  int c0 = t * 2;
  const float* s;
  if (c0 < NF)           s = x   + (size_t)n * NF + c0;
  else if (c0 < NF + HD) s = agg + (size_t)n * HD + (c0 - NF);
  else                   s = u   + (size_t)batch[n] * GD + (c0 - NF - HD);
  out[(size_t)n * 256 + t] = packbf2(s[0], s[1]);
}

__global__ void k_build_Ag(const float* __restrict__ u, const float* __restrict__ xm,
                           unsigned* __restrict__ out) {
  int n = blockIdx.x, t = threadIdx.x;  // 128 thr, 256 cols
  int c0 = t * 2;
  const float* s = (c0 < GD) ? u + (size_t)n * GD + c0
                             : xm + (size_t)n * NF + (c0 - GD);
  out[(size_t)n * 128 + t] = packbf2(s[0], s[1]);
}

__global__ void k_batch_f(const int* __restrict__ b, float* __restrict__ o, int n) {
  int i = blockIdx.x * blockDim.x + threadIdx.x;
  if (i < n) o[i] = (float)b[i];
}

// ---------------------------------------------------------------------------
extern "C" void kernel_launch(void* const* d_in, const int* in_sizes, int n_in,
                              void* d_out, int out_size, void* d_ws, size_t ws_size,
                              hipStream_t stream) {
  (void)in_sizes; (void)out_size; (void)ws_size;
  const int E = E_EDGES, NN = NN_NODES, NG = NG_GRAPHS;
  const float* x0 = (const float*)d_in[0];
  const float* e0 = (const float*)d_in[1];
  const float* u0 = (const float*)d_in[2];
  const int* ei    = (const int*)d_in[n_in - 2];
  const int* batch = (const int*)d_in[n_in - 1];

  char* ws = (char*)d_ws;
  size_t cur = 0;
  auto alloc = [&](size_t bytes) -> void* {
    void* p = ws + cur;
    cur += (bytes + 255) & ~(size_t)255;
    return p;
  };
  unsigned short* A_e  = (unsigned short*)alloc((size_t)E * 512 * 2);
  unsigned short* H_e  = (unsigned short*)alloc((size_t)E * 256 * 2);
  float*          e_buf= (float*)alloc((size_t)E * 128 * 4);
  unsigned short* A_n1 = (unsigned short*)alloc((size_t)E * 256 * 2);
  float*          H1f  = (float*)alloc((size_t)E * 256 * 4);
  unsigned short* H1bf = (unsigned short*)alloc((size_t)E * 256 * 2);
  float*          Hn   = H1f;  // alias: H1f dead after bn_act
  float*          agg  = (float*)alloc((size_t)NN * 256 * 4);
  float*          cntN = (float*)alloc((size_t)NN * 4);
  unsigned short* A_n2 = (unsigned short*)alloc((size_t)NN * 512 * 2);
  float*          H2f  = (float*)alloc((size_t)NN * 256 * 4);
  unsigned short* H2bf = (unsigned short*)alloc((size_t)NN * 256 * 2);
  float*          xA   = (float*)alloc((size_t)NN * 128 * 4);
  float*          xB   = (float*)alloc((size_t)NN * 128 * 4);
  float*          xsum = (float*)alloc((size_t)NG * 128 * 4);
  float*          cntG = (float*)alloc((size_t)NG * 4);
  unsigned short* A_g  = (unsigned short*)alloc((size_t)NG * 256 * 2);
  float*          Hgf  = (float*)alloc((size_t)NG * 256 * 4);
  unsigned short* Hgbf = (unsigned short*)alloc((size_t)NG * 256 * 2);
  float*          uA   = (float*)alloc((size_t)NG * 128 * 4);
  float*          uB   = (float*)alloc((size_t)NG * 128 * 4);
  float*          stats= (float*)alloc(512 * 4);
  unsigned*       Bp   = (unsigned*)alloc((size_t)512 * 256 * 2);

  auto pack = [&](const float* W, int K, int N) {
    int total = (K / 32) * (N / 16) * 32;
    k_pack_w<<<(total + 255) / 256, 256, 0, stream>>>(W, Bp, K, N);
  };
  auto gemm = [&](const unsigned short* A, const float* bias, float* Cf,
                  unsigned short* Cbf, int M, int N, int K, int relu) {
    k_gemm_bf16<<<dim3((M + BM - 1) / BM, N / BN), 256, 0, stream>>>(
        A, Bp, bias, Cf, Cbf, M, N, K, relu);
  };

  const float* xs = x0;
  const float* es = e0;
  const float* us = u0;
  float* xbufs[2] = {xA, xB};
  float* ubufs[2] = {uA, uB};

  for (int L = 0; L < 3; ++L) {
    const float* const* P = (const float* const*)(d_in + 3 + 22 * L);
    float* x_next = xbufs[L & 1];
    float* u_next = ubufs[L & 1];

    // ---- Edge block: Linear(512->256)+ReLU, Linear(256->128)
    k_build_Ae<<<E, 256, 0, stream>>>(xs, es, us, ei, batch, (unsigned*)A_e, E);
    pack(P[0], 512, 256);
    gemm(A_e, P[1], nullptr, H_e, E, 256, 512, 1);
    pack(P[2], 256, 128);
    gemm(H_e, P[3], e_buf, nullptr, E, 128, 256, 0);

    // ---- Node block MLP1: Linear(256->256), BN+ReLU, Linear(256->256)
    k_build_An1<<<E, 128, 0, stream>>>(xs, e_buf, ei, (unsigned*)A_n1, E);
    pack(P[4], 256, 256);
    gemm(A_n1, P[5], H1f, nullptr, E, 256, 256, 0);
    k_zero<<<2, 256, 0, stream>>>(stats, (size_t)512);
    k_colstats<<<(E + 511) / 512, 256, 0, stream>>>(H1f, stats, E, 256, 512);
    k_bn_act_bf16<<<(unsigned)(((size_t)E * 256 + 255) / 256), 256, 0, stream>>>(
        H1f, stats, P[6], P[7], H1bf, E, 256);
    pack(P[8], 256, 256);
    gemm(H1bf, P[9], Hn, nullptr, E, 256, 256, 0);

    // ---- segment-mean over destination nodes
    k_zero<<<(unsigned)(((size_t)NN * 256 + 255) / 256), 256, 0, stream>>>(agg, (size_t)NN * 256);
    k_zero<<<(NN + 255) / 256, 256, 0, stream>>>(cntN, (size_t)NN);
    k_scatter_rows<<<E, 256, 0, stream>>>(Hn, ei + E, agg, cntN, 256);
    k_div_rows<<<NN, 256, 0, stream>>>(agg, cntN, 256);

    // ---- Node block MLP2: Linear(512->256), BN+ReLU, Linear(256->128)
    k_build_An2<<<NN, 256, 0, stream>>>(xs, agg, us, batch, (unsigned*)A_n2);
    pack(P[10], 512, 256);
    gemm(A_n2, P[11], H2f, nullptr, NN, 256, 512, 0);
    k_zero<<<2, 256, 0, stream>>>(stats, (size_t)512);
    k_colstats<<<(NN + 511) / 512, 256, 0, stream>>>(H2f, stats, NN, 256, 512);
    k_bn_act_bf16<<<(unsigned)(((size_t)NN * 256 + 255) / 256), 256, 0, stream>>>(
        H2f, stats, P[12], P[13], H2bf, NN, 256);
    pack(P[14], 256, 128);
    gemm(H2bf, P[15], x_next, nullptr, NN, 128, 256, 0);

    // ---- Global block: mean(x) per graph, Linear(256->256), BN+ReLU, Linear(256->128)
    k_zero<<<(NG * 128 + 255) / 256, 256, 0, stream>>>(xsum, (size_t)NG * 128);
    k_zero<<<1, 256, 0, stream>>>(cntG, (size_t)NG);
    k_scatter_rows<<<NN, 128, 0, stream>>>(x_next, batch, xsum, cntG, 128);
    k_div_rows<<<NG, 128, 0, stream>>>(xsum, cntG, 128);
    k_build_Ag<<<NG, 128, 0, stream>>>(us, xsum, (unsigned*)A_g);
    pack(P[16], 256, 256);
    gemm(A_g, P[17], Hgf, nullptr, NG, 256, 256, 0);
    k_zero<<<2, 256, 0, stream>>>(stats, (size_t)512);
    k_colstats<<<1, 256, 0, stream>>>(Hgf, stats, NG, 256, NG);
    k_bn_act_bf16<<<(NG * 256 + 255) / 256, 256, 0, stream>>>(
        Hgf, stats, P[18], P[19], Hgbf, NG, 256);
    pack(P[20], 256, 128);
    gemm(Hgbf, P[21], u_next, nullptr, NG, 128, 256, 0);

    xs = x_next; es = e_buf; us = u_next;
  }

  // ---- outputs: x, edge_attr, u, batch (flattened, return order)
  float* out = (float*)d_out;
  size_t oX = 0;
  size_t oE = oX + (size_t)NN * 128;
  size_t oU = oE + (size_t)E * 128;
  size_t oB = oU + (size_t)NG * 128;
  hipMemcpyAsync(out + oX, xs, (size_t)NN * 128 * 4, hipMemcpyDeviceToDevice, stream);
  hipMemcpyAsync(out + oE, es, (size_t)E * 128 * 4, hipMemcpyDeviceToDevice, stream);
  hipMemcpyAsync(out + oU, us, (size_t)NG * 128 * 4, hipMemcpyDeviceToDevice, stream);
  k_batch_f<<<(NN + 255) / 256, 256, 0, stream>>>(batch, out + oB, NN);
}
